// LanguageModel_72636486910124
// MI455X (gfx1250) — compile-verified, matrix operand order checked by reading, not run on hardware
//
#include <hip/hip_runtime.h>
#include <hip/hip_bf16.h>

// Problem dims (from reference): B=16, T=32, V=1003, E=100, H=2048
#define BB   16
#define TT   32
#define VV   1003
#define EE   100
#define HH   2048
#define G4   8192    // 4*H
#define EPAD 128     // E padded to multiple of 32 (WMMA K)
#define VPAD 1008    // V padded to multiple of 16 (WMMA N)

#define KSPLIT 8               // waves per LSTM-step block (split-K)
#define KSLICE (HH / KSPLIT)   // 256 K per wave

typedef __bf16 bf16_t;
typedef __bf16 v16bf __attribute__((ext_vector_type(16)));
typedef __bf16 v8bf  __attribute__((ext_vector_type(8)));
typedef float  v8f   __attribute__((ext_vector_type(8)));

// Tensor Data Mover availability (arity differs across toolchains; see bridge doc)
#ifndef HAVE_TDM
#  if defined(__has_builtin)
#    if __has_builtin(__builtin_amdgcn_tensor_load_to_lds) && \
        __has_builtin(__builtin_amdgcn_s_wait_tensorcnt)
#      define HAVE_TDM 1
#    endif
#  endif
#endif
#ifndef HAVE_TDM
#  define HAVE_TDM 0
#endif

#if HAVE_TDM
typedef unsigned int u32x4 __attribute__((ext_vector_type(4)));
typedef int          i32x4 __attribute__((ext_vector_type(4)));
typedef int          i32x8 __attribute__((ext_vector_type(8)));
#endif

// Load one lane's 16 bf16 K-values for a 16x16x32 WMMA tile.
// Per CDNA5 ISA layout: lane holds K = kbase+{0..7} and kbase+16+{0..7}
// (kbase = (lane>>4)*8 folded into the pointer) -> two contiguous 16B loads
// (global_load_b128 from global memory, ds_load_b128 from LDS).
template <typename P>
__device__ __forceinline__ v16bf load_k16(const P* p) {
  union { v16bf v; v8bf h[2]; } u;
  u.h[0] = *reinterpret_cast<const v8bf*>(p);
  u.h[1] = *reinterpret_cast<const v8bf*>(p + 16);
  return u.v;
}

__device__ __forceinline__ float sigmoid_fast(float x) {
  return 1.0f / (1.0f + __expf(-x));
}
__device__ __forceinline__ float tanh_fast(float x) {
  return 2.0f / (1.0f + __expf(-2.0f * x)) - 1.0f;
}

// ---------------------------------------------------------------------------
// Utility fills
// ---------------------------------------------------------------------------
__global__ void fill_f32(float* __restrict__ p, int n, float v) {
  int i = blockIdx.x * blockDim.x + threadIdx.x;
  if (i < n) p[i] = v;
}
__global__ void fill_bf16(bf16_t* __restrict__ p, int n, float v) {
  int i = blockIdx.x * blockDim.x + threadIdx.x;
  if (i < n) p[i] = (bf16_t)v;
}

// ---------------------------------------------------------------------------
// Embedding gather -> bf16, x laid out [T*B, EPAD] (row = t*16 + b), zero pad
// ---------------------------------------------------------------------------
__global__ void embed_kernel(const int* __restrict__ tokens,
                             const float* __restrict__ emb,
                             bf16_t* __restrict__ xpad) {
  int idx = blockIdx.x * blockDim.x + threadIdx.x;
  if (idx >= BB * TT * EPAD) return;
  int e = idx % EPAD;
  int r = idx / EPAD;         // r = t*16 + b
  int t = r / BB;
  int b = r % BB;
  int tok = tokens[b * TT + t];
  float v = (e < EE) ? emb[(size_t)tok * EE + e] : 0.0f;
  xpad[idx] = (bf16_t)v;
}

// ---------------------------------------------------------------------------
// LDS-tiled transpose + fp32->bf16 cast: src [K,N] row-major -> dst [Npad,Kpad]
// grid: (ceil(Kpad/32), ceil(Npad/32)), block (32,8)
// ---------------------------------------------------------------------------
__global__ void transpose_cast_kernel(const float* __restrict__ src,
                                      bf16_t* __restrict__ dst,
                                      int K, int N, int Kpad, int Npad) {
  __shared__ float tile[32][33];
  int k0 = blockIdx.x * 32;
  int n0 = blockIdx.y * 32;
  for (int i = threadIdx.y; i < 32; i += 8) {
    int k = k0 + i, n = n0 + threadIdx.x;            // coalesced read along n
    tile[i][threadIdx.x] = (k < K && n < N) ? src[(size_t)k * N + n] : 0.0f;
  }
  __syncthreads();
  for (int i = threadIdx.y; i < 32; i += 8) {
    int n = n0 + i, k = k0 + threadIdx.x;            // coalesced write along k
    if (n < Npad && k < Kpad) dst[(size_t)n * Kpad + k] = (bf16_t)tile[threadIdx.x][i];
  }
}

// ---------------------------------------------------------------------------
// Generic C = A[M,K](bf16) * Bt[N,K](bf16, pre-transposed) + bias
// One wave per 16x16 output tile, v_wmma_f32_16x16x32_bf16 over K.
// ---------------------------------------------------------------------------
__global__ void gemm_bf16_wmma(const bf16_t* __restrict__ A,
                               const bf16_t* __restrict__ Bt,
                               const float* __restrict__ bias, int biasN,
                               float* __restrict__ Cf, bf16_t* __restrict__ Cbf,
                               int M, int N, int K) {
  const int lane = threadIdx.x & 31;
  const int wave = (blockIdx.x * blockDim.x + threadIdx.x) >> 5;
  const int ntiles = N >> 4;
  const int mt = wave / ntiles;
  const int nt = wave - mt * ntiles;
  if (mt >= (M >> 4)) return;          // wave-uniform: EXEC all-ones for WMMA

  const int lrow  = lane & 15;
  const int kbase = (lane >> 4) << 3;

  const bf16_t* arow = A  + (size_t)(mt * 16 + lrow) * K + kbase;
  const bf16_t* brow = Bt + (size_t)(nt * 16 + lrow) * K + kbase;

  v8f acc = {};
  for (int kc = 0; kc < K; kc += 32) {
    __builtin_prefetch(brow + kc + 256, 0, 2);
    v16bf a = load_k16(arow + kc);
    v16bf b = load_k16(brow + kc);
    acc = __builtin_amdgcn_wmma_f32_16x16x32_bf16(false, a, false, b,
                                                  (short)0, acc, false, false);
  }

  const int n = nt * 16 + lrow;
  float bv = 0.0f;
  if (bias && n < biasN) bv = bias[n];
  const int mbase = mt * 16 + ((lane >> 4) << 3);    // C layout: lanes 16-31 -> M+8
#pragma unroll
  for (int r = 0; r < 8; ++r) {
    float v = acc[r] + bv;
    size_t off = (size_t)(mbase + r) * N + n;
    if (Cf)  Cf[off]  = v;
    if (Cbf) Cbf[off] = (bf16_t)v;
  }
}

// ---------------------------------------------------------------------------
// One masked-LSTM time step, fused and split-K parallelized:
//   - h_prev [16, H] bf16 (64 KiB) staged into LDS once per block via the
//     Tensor Data Mover (tensor_load_to_lds + s_wait_tensorcnt), fallback
//     cooperative copy if the builtin is unavailable.
//   - One block per 16-wide column tile of H; KSPLIT=8 waves each accumulate
//     all 4 gate tiles over a K=256 slice with v_wmma_f32_16x16x32_bf16
//     (A from LDS via ds_load_b128, U from L2 via global_load_b128).
//   - Partials reduced through LDS; wave 0 applies sigm/tanh cell update and
//     the Keras mask carry in registers.
// Grid: H/16 = 128 blocks x 256 threads = 1024 waves per step.
// h ping-pongs across steps (cross-block WAR safety); c[m,n] has one owner.
// ---------------------------------------------------------------------------
__global__ void lstm_step_wmma(const bf16_t* __restrict__ hprev,
                               const bf16_t* __restrict__ Ut,   // [4H, H] bf16
                               const float* __restrict__ xz_t,  // [16, 4H] fp32
                               const int* __restrict__ tokens, int t,
                               float* __restrict__ cst,         // [16, H] fp32
                               bf16_t* __restrict__ hnext,      // [16, H]
                               bf16_t* __restrict__ hs_out) {   // [16, H]
  extern __shared__ char smem_raw[];
  bf16_t* aS  = (bf16_t*)smem_raw;                          // [16][H]  64 KiB
  float*  red = (float*)(smem_raw + (size_t)BB * HH * 2);   // [8][4][32][8] 32 KiB

  const int tid  = threadIdx.x;
  const int lane = tid & 31;
  const int w    = tid >> 5;      // split-K slice id
  const int nt   = blockIdx.x;    // column tile of H

  // ---- Stage h_prev into LDS --------------------------------------------
#if HAVE_TDM
  if (tid == 0) {
    // D# per CDNA5 ISA: 2D tensor, tile == tensor,
    // dim0 = H (contiguous), dim1 = 16 rows, data_size = 2 bytes.
    unsigned long long ga = (unsigned long long)(uintptr_t)hprev;
    unsigned lds_off = (unsigned)(uintptr_t)aS;
    u32x4 g0;
    g0.x = 1u;                                        // count=1 (valid user D#)
    g0.y = lds_off;                                   // lds_addr
    g0.z = (unsigned)(ga & 0xFFFFFFFFu);              // global_addr[31:0]
    g0.w = (unsigned)((ga >> 32) & 0x1FFFFFFu)        // global_addr[56:32]
         | (2u << 30);                                // type = 2 (image)
    i32x8 g1;
    g1[0] = (int)(1u << 16);                          // data_size = 2B
    g1[1] = (int)(((unsigned)HH & 0xFFFFu) << 16);    // tensor_dim0[15:0]
    g1[2] = (int)(((unsigned)HH >> 16) |
                  (((unsigned)BB & 0xFFFFu) << 16));  // dim0[31:16] | dim1[15:0]
    g1[3] = (int)(((unsigned)BB >> 16) |
                  (((unsigned)HH & 0xFFFFu) << 16));  // dim1[31:16] | tile_dim0
    g1[4] = (int)((unsigned)BB & 0xFFFFu);            // tile_dim1=16, tile_dim2=0
    g1[5] = (int)(unsigned)HH;                        // dim0_stride[31:0]
    g1[6] = (int)(((unsigned)HH & 0xFFFFu) << 16);    // stride0[47:32]=0 | stride1[15:0]
    g1[7] = (int)((unsigned)HH >> 16);                // stride1[47:16]
    i32x4 gz = {0, 0, 0, 0};
#if defined(__clang_major__) && (__clang_major__ >= 23)
    i32x8 gz8 = {0, 0, 0, 0, 0, 0, 0, 0};
    __builtin_amdgcn_tensor_load_to_lds(g0, g1, gz, gz, gz8, 0);
#else
    __builtin_amdgcn_tensor_load_to_lds(g0, g1, gz, gz, 0);
#endif
  }
  if (tid < 32) __builtin_amdgcn_s_wait_tensorcnt(0);  // issuing wave drains TENSORcnt
  __syncthreads();
#else
  for (int i = tid; i < (BB * HH) / 8; i += blockDim.x)
    ((v8bf*)aS)[i] = ((const v8bf*)hprev)[i];
  __syncthreads();
#endif

  // ---- Split-K WMMA accumulation ----------------------------------------
  const int lrow  = lane & 15;
  const int kbase = (lane >> 4) << 3;
  const int k0    = w * KSLICE;

  const bf16_t* arow = aS + (size_t)lrow * HH + k0 + kbase;
  const bf16_t* b0 = Ut + (size_t)(0 * HH + nt * 16 + lrow) * HH + k0 + kbase;
  const bf16_t* b1 = Ut + (size_t)(1 * HH + nt * 16 + lrow) * HH + k0 + kbase;
  const bf16_t* b2 = Ut + (size_t)(2 * HH + nt * 16 + lrow) * HH + k0 + kbase;
  const bf16_t* b3 = Ut + (size_t)(3 * HH + nt * 16 + lrow) * HH + k0 + kbase;

  v8f acc0 = {}, acc1 = {}, acc2 = {}, acc3 = {};
  for (int kc = 0; kc < KSLICE; kc += 32) {
    v16bf a = load_k16(arow + kc);
    acc0 = __builtin_amdgcn_wmma_f32_16x16x32_bf16(false, a, false, load_k16(b0 + kc),
                                                   (short)0, acc0, false, false);
    acc1 = __builtin_amdgcn_wmma_f32_16x16x32_bf16(false, a, false, load_k16(b1 + kc),
                                                   (short)0, acc1, false, false);
    acc2 = __builtin_amdgcn_wmma_f32_16x16x32_bf16(false, a, false, load_k16(b2 + kc),
                                                   (short)0, acc2, false, false);
    acc3 = __builtin_amdgcn_wmma_f32_16x16x32_bf16(false, a, false, load_k16(b3 + kc),
                                                   (short)0, acc3, false, false);
  }

  // ---- LDS reduction of split-K partials --------------------------------
  {
    float* base = red + (size_t)w * 1024 + (size_t)lane * 8;  // [w][gate][lane][8]
    *(v8f*)(base + 0 * 256) = acc0;
    *(v8f*)(base + 1 * 256) = acc1;
    *(v8f*)(base + 2 * 256) = acc2;
    *(v8f*)(base + 3 * 256) = acc3;
  }
  __syncthreads();
  if (w != 0) return;

  v8f s0 = {}, s1 = {}, s2 = {}, s3 = {};
#pragma unroll
  for (int ww = 0; ww < KSPLIT; ++ww) {
    const float* b = red + (size_t)ww * 1024 + (size_t)lane * 8;
    s0 += *(const v8f*)(b + 0 * 256);
    s1 += *(const v8f*)(b + 1 * 256);
    s2 += *(const v8f*)(b + 2 * 256);
    s3 += *(const v8f*)(b + 3 * 256);
  }

  // ---- Fused LSTM cell update + Keras masking ---------------------------
  const int n     = nt * 16 + lrow;
  const int mbase = (lane >> 4) << 3;
#pragma unroll
  for (int r = 0; r < 8; ++r) {
    const int m = mbase + r;                   // batch index
    const float* xzr = xz_t + (size_t)m * G4;
    float zi = xzr[0 * HH + n] + s0[r];
    float zf = xzr[1 * HH + n] + s1[r];
    float zg = xzr[2 * HH + n] + s2[r];
    float zo = xzr[3 * HH + n] + s3[r];
    float ig = sigmoid_fast(zi);
    float fg = sigmoid_fast(zf);
    float gg = tanh_fast(zg);
    float og = sigmoid_fast(zo);
    float cold = cst[(size_t)m * HH + n];
    float cnew = fg * cold + ig * gg;
    float hnew = og * tanh_fast(cnew);
    if (tokens[m * TT + t] == 0) {             // masked step: carry h,c through
      cnew = cold;
      hnew = (float)aS[(size_t)m * HH + n];    // h_prev from LDS stage
    }
    cst[(size_t)m * HH + n] = cnew;
    bf16_t hb = (bf16_t)hnew;
    hnext[(size_t)m * HH + n]  = hb;
    hs_out[(size_t)m * HH + n] = hb;
  }
}

// ---------------------------------------------------------------------------
// Row softmax over the first VV of VPAD logits; 16 rows, one block per row.
// ---------------------------------------------------------------------------
__global__ void softmax_kernel(const float* __restrict__ logits,
                               float* __restrict__ out) {
  __shared__ float red[256];
  const int row = blockIdx.x;
  const float* lr = logits + (size_t)row * VPAD;
  float mx = -3.4e38f;
  for (int i = threadIdx.x; i < VV; i += blockDim.x) mx = fmaxf(mx, lr[i]);
  red[threadIdx.x] = mx; __syncthreads();
  for (int s = 128; s > 0; s >>= 1) {
    if (threadIdx.x < s) red[threadIdx.x] = fmaxf(red[threadIdx.x], red[threadIdx.x + s]);
    __syncthreads();
  }
  mx = red[0]; __syncthreads();
  float sum = 0.0f;
  for (int i = threadIdx.x; i < VV; i += blockDim.x) sum += __expf(lr[i] - mx);
  red[threadIdx.x] = sum; __syncthreads();
  for (int s = 128; s > 0; s >>= 1) {
    if (threadIdx.x < s) red[threadIdx.x] += red[threadIdx.x + s];
    __syncthreads();
  }
  const float inv = 1.0f / red[0];
  for (int i = threadIdx.x; i < VV; i += blockDim.x)
    out[(size_t)row * VV + i] = __expf(lr[i] - mx) * inv;
}

// ---------------------------------------------------------------------------
// Host orchestration
// ---------------------------------------------------------------------------
extern "C" void kernel_launch(void* const* d_in, const int* in_sizes, int n_in,
                              void* d_out, int out_size, void* d_ws, size_t ws_size,
                              hipStream_t stream) {
  (void)in_sizes; (void)n_in; (void)out_size; (void)ws_size;
  const int*   tokens = (const int*)  d_in[0];
  const float* emb    = (const float*)d_in[1];
  const float* Wl[3]  = { (const float*)d_in[2], (const float*)d_in[5], (const float*)d_in[8] };
  const float* Ul[3]  = { (const float*)d_in[3], (const float*)d_in[6], (const float*)d_in[9] };
  const float* bl[3]  = { (const float*)d_in[4], (const float*)d_in[7], (const float*)d_in[10] };
  const float* Wd1 = (const float*)d_in[11];
  const float* bd1 = (const float*)d_in[12];
  const float* Wd2 = (const float*)d_in[13];
  const float* bd2 = (const float*)d_in[14];
  float* out = (float*)d_out;

  // Workspace carving (~85 MiB total; per-layer Wt/Ut buffers are reused).
  char* ws = (char*)d_ws;
  size_t off = 0;
  auto carve = [&](size_t bytes) -> char* {
    char* p = ws + off;
    off = (off + bytes + 255) & ~(size_t)255;
    return p;
  };
  bf16_t* bufW  = (bf16_t*)carve((size_t)G4 * HH * 2);       // 32 MiB (max use)
  bf16_t* bufU  = (bf16_t*)carve((size_t)G4 * HH * 2);       // 32 MiB
  float*  xz    = (float*) carve((size_t)BB * TT * G4 * 4);  // 16 MiB
  bf16_t* hsA   = (bf16_t*)carve((size_t)BB * TT * HH * 2);  // 2 MiB
  bf16_t* hsB   = (bf16_t*)carve((size_t)BB * TT * HH * 2);  // 2 MiB
  bf16_t* xpad  = (bf16_t*)carve((size_t)BB * TT * EPAD * 2);
  bf16_t* hping = (bf16_t*)carve((size_t)BB * HH * 2);
  bf16_t* hpong = (bf16_t*)carve((size_t)BB * HH * 2);
  float*  cbuf  = (float*) carve((size_t)BB * HH * 4);
  bf16_t* ybf   = (bf16_t*)carve((size_t)BB * HH * 2);
  float*  lg    = (float*) carve((size_t)BB * VPAD * 4);

  const dim3 tb(32, 8);
  const size_t lstm_smem = (size_t)BB * HH * 2            // staged h_prev
                         + (size_t)KSPLIT * 4 * 32 * 8 * 4; // split-K partials

  // 1) embedding gather -> bf16, rows = t*16 + b
  {
    int n = BB * TT * EPAD;
    embed_kernel<<<(n + 255) / 256, 256, 0, stream>>>(tokens, emb, xpad);
  }

  bf16_t* layer_in[3]    = { xpad, hsA, hsB };
  int     layer_K[3]     = { EPAD, HH, HH };
  int     layer_Kreal[3] = { EE, HH, HH };
  bf16_t* layer_hs[3]    = { hsA, hsB, hsA };  // layer-3 hs is dead output
  bf16_t* hbuf[2]        = { hping, hpong };

  for (int l = 0; l < 3; ++l) {
    // Weight prep: W [Kreal, 4H] -> bufW [4H, Kpad]; U [H, 4H] -> bufU [4H, H]
    {
      dim3 grid((layer_K[l] + 31) / 32, (G4 + 31) / 32);
      transpose_cast_kernel<<<grid, tb, 0, stream>>>(Wl[l], bufW,
                                                     layer_Kreal[l], G4, layer_K[l], G4);
    }
    {
      dim3 grid((HH + 31) / 32, (G4 + 31) / 32);
      transpose_cast_kernel<<<grid, tb, 0, stream>>>(Ul[l], bufU, HH, G4, HH, G4);
    }
    // xz = in @ W + b : M=512, N=8192, K=layer_K  (fully parallel WMMA GEMM)
    {
      int tiles   = (BB * TT / 16) * (G4 / 16);
      int threads = tiles * 32;
      gemm_bf16_wmma<<<(threads + 255) / 256, 256, 0, stream>>>(
          layer_in[l], bufW, bl[l], G4, xz, (bf16_t*)nullptr, BB * TT, G4, layer_K[l]);
    }
    // h0 = 0, c0 = 0
    {
      int n = BB * HH;
      fill_bf16<<<(n + 255) / 256, 256, 0, stream>>>(hbuf[0], n, 0.0f);
      fill_f32 <<<(n + 255) / 256, 256, 0, stream>>>(cbuf, n, 0.0f);
    }
    // Sequential recurrence: fused TDM-staged, split-K WMMA step kernel.
    // U (32 MiB bf16) stays resident in the 192 MB L2 across all 32 steps.
    for (int t = 0; t < TT; ++t) {
      const bf16_t* hp = hbuf[t & 1];
      bf16_t*       hn = hbuf[(t + 1) & 1];
      lstm_step_wmma<<<HH / 16, KSPLIT * 32, lstm_smem, stream>>>(
          hp, bufU, xz + (size_t)t * BB * G4, tokens, t, cbuf, hn,
          layer_hs[l] + (size_t)t * BB * HH);
    }
    // T=32 is even -> h_last lands in hbuf[0]
  }

  // Head: y = h_last @ Wd1 + bd1 (bf16 out), logits = y @ Wd2 + bd2 (fp32)
  {
    dim3 grid((HH + 31) / 32, (HH + 31) / 32);
    transpose_cast_kernel<<<grid, tb, 0, stream>>>(Wd1, bufW, HH, HH, HH, HH);
    int threads = (BB / 16) * (HH / 16) * 32;
    gemm_bf16_wmma<<<(threads + 255) / 256, 256, 0, stream>>>(
        hbuf[0], bufW, bd1, HH, (float*)nullptr, ybf, BB, HH, HH);
  }
  {
    dim3 grid((HH + 31) / 32, (VPAD + 31) / 32);
    transpose_cast_kernel<<<grid, tb, 0, stream>>>(Wd2, bufW, HH, VV, HH, VPAD);
    int threads = (BB / 16) * (VPAD / 16) * 32;
    gemm_bf16_wmma<<<(threads + 255) / 256, 256, 0, stream>>>(
        ybf, bufW, bd2, VV, lg, (bf16_t*)nullptr, BB, VPAD, HH);
  }

  softmax_kernel<<<BB, 256, 0, stream>>>(lg, out);
}
